// MultiStrategyController_77549929496723
// MI455X (gfx1250) — compile-verified, matrix operand order checked by reading, not run on hardware
//
#include <hip/hip_runtime.h>
#include <hip/hip_bf16.h>

typedef __attribute__((ext_vector_type(16))) _Float16 v16h;
typedef __attribute__((ext_vector_type(8)))  _Float16 v8h;
typedef __attribute__((ext_vector_type(8)))  float    v8f;

#define BATCH     131072
#define STATE_DIM 256
#define XCOLS     260   // STATE_DIM + NUM_GOALS
#define HID       256
#define HEAD_HID  128
#define OUT_DIM   128
#define NUM_GOALS 4
#define LN_EPS    1e-5f

#define WAVES_PER_BLOCK 4
#define ROWS_PER_WAVE   16
#define ROWS_PER_BLOCK  (WAVES_PER_BLOCK * ROWS_PER_WAVE)

// padded LDS strides (halves); both are multiples of 8 halves (16B) for b128 loads
#define XS_STRIDE 264   // HID + 8
#define TS_STRIDE 136   // HEAD_HID + 8

// fragment-packed weight sizes (halves)
#define W0F_ELEMS (16 * 8 * 32 * 16)      // 16 ntiles x 8 kchunks x 32 lanes x 16 halves
#define W1F_ELEMS (NUM_GOALS * 8 * 8 * 32 * 16)
#define W2F_ELEMS (NUM_GOALS * 8 * 4 * 32 * 16)

// ---------------------------------------------------------------------------
// Weight pre-pack: row-major f32 W[g][k][n] -> per-lane WMMA B-fragment f16.
// 16-bit 32x16 B fragment layout (mirror of the documented 16x32 A layout):
//   lane holds column n = j*16 + (lane&15);
//   half i maps to K = c*32 + (i&7) + (i&8 ? 16:0) + (lane&16 ? 8:0)
// Output element t is stored exactly at its consumption address:
//   t = ((((g*NT + j)*NC + c) * 32 + lane) * 16) + i
// ---------------------------------------------------------------------------
__global__ void prep_wfrag(const float* __restrict__ W, _Float16* __restrict__ out,
                           int NT, int NC, int N, int total) {
    int t = blockIdx.x * blockDim.x + threadIdx.x;
    if (t >= total) return;
    int i    = t & 15;
    int lane = (t >> 4) & 31;
    int f    = t >> 9;
    int c    = f % NC;
    int jg   = f / NC;
    int j    = jg % NT;
    int g    = jg / NT;
    int K    = NC * 32;
    int k    = c * 32 + (i & 7) + ((i & 8) ? 16 : 0) + ((lane & 16) ? 8 : 0);
    int n    = j * 16 + (lane & 15);
    out[t] = (_Float16)W[((size_t)g * K + k) * N + n];
}

// Gather a 16x32 f16 A-fragment from a row-major padded LDS tile.
// lane holds row m = lane&15; halves i<8 are K = base..base+7 (one b128),
// halves i>=8 are K = base+16..base+23 (second b128).
__device__ __forceinline__ v16h afrag(const _Float16* base, int stride, int cc, int lane) {
    int m   = lane & 15;
    int off = m * stride + cc * 32 + ((lane & 16) ? 8 : 0);
    v8h lo = *(const v8h*)(base + off);
    v8h hi = *(const v8h*)(base + off + 16);
    return __builtin_shufflevector(lo, hi, 0, 1, 2, 3, 4, 5, 6, 7,
                                           8, 9, 10, 11, 12, 13, 14, 15);
}

__device__ __forceinline__ v8f wmma_f16(v16h a, v16h b, v8f c) {
    return __builtin_amdgcn_wmma_f32_16x16x32_f16(false, a, false, b,
                                                  (short)0, c, false, false);
}

// ---------------------------------------------------------------------------
// Fused controller kernel: 4 waves/block, 16 rows/wave.
// ---------------------------------------------------------------------------
__global__ void __launch_bounds__(WAVES_PER_BLOCK * 32)
controller_kernel(const float* __restrict__ x,
                  const float* __restrict__ b0,
                  const float* __restrict__ gamma,
                  const float* __restrict__ beta,
                  const float* __restrict__ b1,
                  const float* __restrict__ b2,
                  const _Float16* __restrict__ w0f,
                  const _Float16* __restrict__ w1f,
                  const _Float16* __restrict__ w2f,
                  float* __restrict__ out) {
    __shared__ _Float16 xs[WAVES_PER_BLOCK][ROWS_PER_WAVE * XS_STRIDE];
    __shared__ _Float16 ts[WAVES_PER_BLOCK][ROWS_PER_WAVE * TS_STRIDE];
    __shared__ int      gidx[WAVES_PER_BLOCK][ROWS_PER_WAVE];

    const int lane = threadIdx.x & 31;
    const int wid  = threadIdx.x >> 5;
    const int rowbase = blockIdx.x * ROWS_PER_BLOCK + wid * ROWS_PER_WAVE;
    const int nlane = lane & 15;          // column within 16-wide N tile
    const int mhalf = (lane >> 4) << 3;   // row offset for C/D fragment halves

    // ---- stage state tile (f32 -> f16) into LDS, coalesced float4 loads ----
    #pragma unroll
    for (int r = 0; r < ROWS_PER_WAVE; ++r) {
        const float* xrow = x + (size_t)(rowbase + r) * XCOLS;
        #pragma unroll
        for (int it = 0; it < 2; ++it) {
            int cbase = (it * 32 + lane) * 4;
            float4 v = *(const float4*)(xrow + cbase);
            _Float16* dst = &xs[wid][r * XS_STRIDE + cbase];
            dst[0] = (_Float16)v.x; dst[1] = (_Float16)v.y;
            dst[2] = (_Float16)v.z; dst[3] = (_Float16)v.w;
        }
    }
    // ---- per-row goal argmax (first-max wins, as jnp.argmax) ----
    if (lane < ROWS_PER_WAVE) {
        const float* gp = x + (size_t)(rowbase + lane) * XCOLS + STATE_DIM;
        float bv = gp[0]; int bi = 0;
        #pragma unroll
        for (int g = 1; g < NUM_GOALS; ++g) {
            float v = gp[g];
            if (v > bv) { bv = v; bi = g; }
        }
        gidx[wid][lane] = bi;
    }
    __syncthreads();

    // ---- layer 0: h = state @ W0 + b0  (16 N-tiles, K = 256 = 8 chunks) ----
    v8f hacc[16];
    #pragma unroll
    for (int j = 0; j < 16; ++j) {
        v8f c = {};
        #pragma unroll
        for (int cc = 0; cc < 8; ++cc) {
            v16h a = afrag(xs[wid], XS_STRIDE, cc, lane);
            v16h b = *(const v16h*)(w0f + (((j * 8 + cc) * 32 + lane) << 4));
            c = wmma_f16(a, b, c);
        }
        float bias = b0[j * 16 + nlane];
        #pragma unroll
        for (int r = 0; r < 8; ++r) c[r] += bias;
        hacc[j] = c;
    }

    // ---- LayerNorm row stats via shfl-xor butterflies over the 16-lane group ----
    float mu[8], rstd[8];
    #pragma unroll
    for (int r = 0; r < 8; ++r) {
        float s1 = 0.f, s2 = 0.f;
        #pragma unroll
        for (int j = 0; j < 16; ++j) { float v = hacc[j][r]; s1 += v; s2 += v * v; }
        #pragma unroll
        for (int m = 8; m >= 1; m >>= 1) {
            s1 += __shfl_xor(s1, m, 32);
            s2 += __shfl_xor(s2, m, 32);
        }
        float mean = s1 * (1.0f / HID);
        float var  = s2 * (1.0f / HID) - mean * mean;
        mu[r]   = mean;
        rstd[r] = rsqrtf(var + LN_EPS);
    }

    // ---- normalize + leaky ReLU, write h (f16) back over the x tile ----
    #pragma unroll
    for (int j = 0; j < 16; ++j) {
        int n = j * 16 + nlane;
        float ga = gamma[n], be = beta[n];
        #pragma unroll
        for (int r = 0; r < 8; ++r) {
            float v = (hacc[j][r] - mu[r]) * rstd[r] * ga + be;
            v = (v >= 0.f) ? v : 0.1f * v;
            int m = r + mhalf;
            xs[wid][m * XS_STRIDE + n] = (_Float16)v;
        }
    }
    __syncthreads();

    // ---- per-goal heads (all 4 computed; select at the store) ----
    for (int g = 0; g < NUM_GOALS; ++g) {
        // layer 1: t = relu(h @ W1[g] + b1[g]),  8 N-tiles, K = 256
        #pragma unroll
        for (int j = 0; j < 8; ++j) {
            v8f c = {};
            #pragma unroll
            for (int cc = 0; cc < 8; ++cc) {
                v16h a = afrag(xs[wid], XS_STRIDE, cc, lane);
                v16h b = *(const v16h*)(w1f + ((((g * 8 + j) * 8 + cc) * 32 + lane) << 4));
                c = wmma_f16(a, b, c);
            }
            float bias = b1[g * HEAD_HID + j * 16 + nlane];
            #pragma unroll
            for (int r = 0; r < 8; ++r) {
                float v = c[r] + bias;
                v = (v > 0.f) ? v : 0.f;
                int m = r + mhalf;
                ts[wid][m * TS_STRIDE + j * 16 + nlane] = (_Float16)v;
            }
        }
        __syncthreads();

        // layer 2: o = t @ W2[g] + b2[g],  8 N-tiles, K = 128 = 4 chunks
        #pragma unroll
        for (int j = 0; j < 8; ++j) {
            v8f c = {};
            #pragma unroll
            for (int cc = 0; cc < 4; ++cc) {
                v16h a = afrag(ts[wid], TS_STRIDE, cc, lane);
                v16h b = *(const v16h*)(w2f + ((((g * 8 + j) * 4 + cc) * 32 + lane) << 4));
                c = wmma_f16(a, b, c);
            }
            float bias = b2[g * OUT_DIM + j * 16 + nlane];
            #pragma unroll
            for (int r = 0; r < 8; ++r) {
                int m = r + mhalf;
                if (gidx[wid][m] == g) {
                    out[(size_t)(rowbase + m) * OUT_DIM + j * 16 + nlane] = c[r] + bias;
                }
            }
        }
        __syncthreads();
    }
}

extern "C" void kernel_launch(void* const* d_in, const int* in_sizes, int n_in,
                              void* d_out, int out_size, void* d_ws, size_t ws_size,
                              hipStream_t stream) {
    const float* x     = (const float*)d_in[0];
    const float* W0    = (const float*)d_in[1];
    const float* b0    = (const float*)d_in[2];
    const float* gamma = (const float*)d_in[3];
    const float* beta  = (const float*)d_in[4];
    const float* W1    = (const float*)d_in[5];
    const float* b1    = (const float*)d_in[6];
    const float* W2    = (const float*)d_in[7];
    const float* b2    = (const float*)d_in[8];
    float* out = (float*)d_out;

    _Float16* w0f = (_Float16*)d_ws;
    _Float16* w1f = w0f + W0F_ELEMS;
    _Float16* w2f = w1f + W1F_ELEMS;

    // pre-pack weights into WMMA B-fragment layout (cheap; L2-resident thereafter)
    prep_wfrag<<<(W0F_ELEMS + 255) / 256, 256, 0, stream>>>(W0, w0f, 16, 8, HID,      W0F_ELEMS);
    prep_wfrag<<<(W1F_ELEMS + 255) / 256, 256, 0, stream>>>(W1, w1f,  8, 8, HEAD_HID, W1F_ELEMS);
    prep_wfrag<<<(W2F_ELEMS + 255) / 256, 256, 0, stream>>>(W2, w2f,  8, 4, OUT_DIM,  W2F_ELEMS);

    dim3 grid(BATCH / ROWS_PER_BLOCK);
    dim3 block(WAVES_PER_BLOCK * 32);
    controller_kernel<<<grid, block, 0, stream>>>(x, b0, gamma, beta, b1, b2,
                                                  w0f, w1f, w2f, out);
}
